// PanODE_68736656605347
// MI455X (gfx1250) — compile-verified, matrix operand order checked by reading, not run on hardware
//
#include <hip/hip_runtime.h>
#include <hip/hip_bf16.h>
#include <math.h>

// ---------------------------------------------------------------------------
// Neural-ODE RK4 solver, MI455X (gfx1250, wave32, WMMA).
// traj[t+1] = RK4 step of f(y) = tanh(y@W1+b1)@W2+b2.
// GEMMs: v_wmma_f32_16x16x32_f16, double-buffered LDS fed by
// global_load_async_to_lds_b128 (ASYNCcnt), branch-free tanh on TRANS units.
// ---------------------------------------------------------------------------

typedef __attribute__((ext_vector_type(16))) _Float16 v16h;
typedef __attribute__((ext_vector_type(8)))  float    v8f;

#define BATCH  1024
#define DIM    1024
#define HIDDEN 4096
#define TSTEPS 16
#define BD     (BATCH * DIM)

#define BKK     32            // K-slab per LDS stage
#define LDSS    (BKK + 8)     // padded LDS row stride (halves): 80B = 20 banks

union FragU { v16h h; uint4 u[2]; };

// tanh(x) = 1 - 2/(exp2(2*log2e*x)+1): branch-free, saturates to +-1,
// propagates NaN; v_exp_f32 / v_rcp_f32 are TRANS ops that co-execute
// with the WMMA pipeline (unlike libm tanhf's branchy expansion).
__device__ __forceinline__ float fast_tanh(float x) {
  const float e = __builtin_amdgcn_exp2f(x * 2.885390081777926815f);
  return fmaf(-2.0f, __builtin_amdgcn_rcpf(e + 1.0f), 1.0f);
}

// ---------------------------------------------------------------------------
// Tiled WMMA GEMM:  out = act(A[M][K] * Bt[N][K]^T + bias)
//   A  : f16 row-major [M][K]
//   Bt : f16 row-major [N][K]  (pre-transposed weights -> B-fragments load
//                               exactly like A-fragments: contiguous K runs)
// 256 threads = 8 wave32s, wave grid 4(M) x 2(N). LDS double-buffered,
// filled by async DMA while WMMAs consume the other buffer.
// ---------------------------------------------------------------------------
template <int BM, int BN, int WTM, int WTN, bool TANH_F16>
__global__ __launch_bounds__(256) void gemm_wmma_kernel(
    const _Float16* __restrict__ A,
    const _Float16* __restrict__ Bt,
    const float*    __restrict__ bias,
    void*           __restrict__ outp,
    int M, int N, int K) {
  static_assert((BM / (WTM * 16)) * (BN / (WTN * 16)) == 8, "need 8 waves");
  static_assert(BN / (WTN * 16) == 2, "wave grid is Mx2");
  constexpr int A_VECS = (BM * BKK) / (256 * 8);  // 16B chunks per thread
  constexpr int B_VECS = (BN * BKK) / (256 * 8);
  constexpr int ATILE  = BM * LDSS;               // halves per buffer
  constexpr int BTILE  = BN * LDSS;

  __shared__ _Float16 sA[2 * ATILE];
  __shared__ _Float16 sB[2 * BTILE];

  const int tid    = threadIdx.x;
  const int wave   = tid >> 5;
  const int lane   = tid & 31;
  const int lane_r = lane & 15;   // row inside 16-wide fragment
  const int lane_h = lane >> 4;   // which K-half this lane holds

  const int m0 = blockIdx.y * BM;
  const int n0 = blockIdx.x * BN;
  const int wm = (wave >> 1) * (WTM * 16);
  const int wn = (wave & 1) * (WTN * 16);

  v8f acc[WTM][WTN] = {};

  // async DMA one K-slab (global -> LDS buffer `buf`), tracked by ASYNCcnt
  auto issue_tile = [&](int k0, int buf) {
#pragma unroll
    for (int v = 0; v < A_VECS; ++v) {
      const int h = (tid + v * 256) * 8, r = h >> 5, c = h & 31;
      const _Float16* g = A + (size_t)(m0 + r) * K + (k0 + c);
      const unsigned  l =
          (unsigned)(uintptr_t)&sA[buf * ATILE + r * LDSS + c];
      asm volatile("global_load_async_to_lds_b128 %0, %1, off"
                   :: "v"(l), "v"(g) : "memory");
    }
#pragma unroll
    for (int v = 0; v < B_VECS; ++v) {
      const int h = (tid + v * 256) * 8, r = h >> 5, c = h & 31;
      const _Float16* g = Bt + (size_t)(n0 + r) * K + (k0 + c);
      const unsigned  l =
          (unsigned)(uintptr_t)&sB[buf * BTILE + r * LDSS + c];
      asm volatile("global_load_async_to_lds_b128 %0, %1, off"
                   :: "v"(l), "v"(g) : "memory");
    }
  };

  issue_tile(0, 0);
  asm volatile("s_wait_asynccnt 0x0" ::: "memory");
  __syncthreads();

  int cur = 0;
  for (int k0 = 0; k0 < K; k0 += BKK) {
    // start DMA of the next slab into the other buffer; overlaps the WMMAs
    if (k0 + BKK < K) issue_tile(k0 + BKK, cur ^ 1);
    if (k0 + 2 * BKK < K) {  // pull slab after next into L2 (global_prefetch_b8)
      __builtin_prefetch(A + (size_t)(m0 + (tid >> 1)) * K + (k0 + 2 * BKK), 0, 1);
      __builtin_prefetch(Bt + (size_t)(n0 + (tid >> 1)) * K + (k0 + 2 * BKK), 0, 1);
    }

    // fragments: lane = row, two contiguous 8-half runs per 16-bit ISA layout
    v16h afrag[WTM], bfrag[WTN];
#pragma unroll
    for (int i = 0; i < WTM; ++i) {
      const _Float16* p = &sA[cur * ATILE + (wm + i * 16 + lane_r) * LDSS];
      FragU f;
      f.u[0] = *(const uint4*)(p + lane_h * 8);        // K = h*8 .. h*8+7
      f.u[1] = *(const uint4*)(p + 16 + lane_h * 8);   // K = 16+h*8 ..
      afrag[i] = f.h;
    }
#pragma unroll
    for (int j = 0; j < WTN; ++j) {
      const _Float16* p = &sB[cur * BTILE + (wn + j * 16 + lane_r) * LDSS];
      FragU f;
      f.u[0] = *(const uint4*)(p + lane_h * 8);
      f.u[1] = *(const uint4*)(p + 16 + lane_h * 8);
      bfrag[j] = f.h;
    }
#pragma unroll
    for (int i = 0; i < WTM; ++i)
#pragma unroll
      for (int j = 0; j < WTN; ++j)
        acc[i][j] = __builtin_amdgcn_wmma_f32_16x16x32_f16(
            false, afrag[i], false, bfrag[j], (short)0, acc[i][j], false, false);

    asm volatile("s_wait_asynccnt 0x0" ::: "memory");  // next buffer is filled
    __syncthreads();
    cur ^= 1;
  }

  // epilogue: C layout — VGPR g: M = (lane>>4)*8 + g, N = lane&15
#pragma unroll
  for (int j = 0; j < WTN; ++j) {
    const int   n  = n0 + wn + j * 16 + lane_r;
    const float bv = bias[n];
#pragma unroll
    for (int i = 0; i < WTM; ++i) {
      const int mbase = m0 + wm + i * 16 + lane_h * 8;
#pragma unroll
      for (int g = 0; g < 8; ++g) {
        const float v = acc[i][j][g] + bv;
        if constexpr (TANH_F16) {
          ((_Float16*)outp)[(size_t)(mbase + g) * N + n] = (_Float16)fast_tanh(v);
        } else {
          ((float*)outp)[(size_t)(mbase + g) * N + n] = v;
        }
      }
    }
  }
}

// W [K][N] f32 row-major  ->  Wt [N][K] f16 (one-time, L2-resident, negligible)
__global__ void prep_weight_kernel(const float* __restrict__ W,
                                   _Float16* __restrict__ Wt, int K, int N) {
  size_t idx = (size_t)blockIdx.x * blockDim.x + threadIdx.x;
  if (idx >= (size_t)K * N) return;
  const int n = (int)(idx / K);
  const int k = (int)(idx % K);
  Wt[idx] = (_Float16)W[(size_t)k * N + n];
}

// d_out[0..15] = t_span; traj[0] = y_init; yh = f16(y_init)
__global__ void init_kernel(const float* __restrict__ y_init,
                            const float* __restrict__ t_span,
                            float* __restrict__ out,
                            _Float16* __restrict__ yh) {
  const int i = blockIdx.x * blockDim.x + threadIdx.x;
  if (i < TSTEPS) out[i] = t_span[i];
  if (i < BD) {
    const float v = y_init[i];
    out[TSTEPS + i] = v;
    yh[i] = (_Float16)v;
  }
}

// yh = f16(y + coef*h*k)   (h read from t_span on-device: graph-capture safe)
__global__ void stage_axpy_kernel(const float* __restrict__ y,
                                  const float* __restrict__ k,
                                  const float* __restrict__ t_span, int step,
                                  float coef, _Float16* __restrict__ yh) {
  const int i = blockIdx.x * blockDim.x + threadIdx.x;
  if (i < BD) {
    const float h = t_span[step + 1] - t_span[step];
    yh[i] = (_Float16)(y[i] + coef * h * k[i]);
  }
}

// ynext = y + h/6*(k1+2k2+2k3+k4); yh = f16(ynext)
__global__ void rk4_update_kernel(const float* __restrict__ y,
                                  const float* __restrict__ k1,
                                  const float* __restrict__ k2,
                                  const float* __restrict__ k3,
                                  const float* __restrict__ k4,
                                  const float* __restrict__ t_span, int step,
                                  float* __restrict__ ynext,
                                  _Float16* __restrict__ yh) {
  const int i = blockIdx.x * blockDim.x + threadIdx.x;
  if (i < BD) {
    const float h = t_span[step + 1] - t_span[step];
    const float v =
        y[i] + (h * (1.0f / 6.0f)) * (k1[i] + 2.0f * (k2[i] + k3[i]) + k4[i]);
    ynext[i] = v;
    yh[i] = (_Float16)v;
  }
}

extern "C" void kernel_launch(void* const* d_in, const int* in_sizes, int n_in,
                              void* d_out, int out_size, void* d_ws,
                              size_t ws_size, hipStream_t stream) {
  (void)in_sizes; (void)n_in; (void)out_size; (void)ws_size;
  const float* y_init = (const float*)d_in[0];
  const float* t_span = (const float*)d_in[1];
  const float* W1     = (const float*)d_in[2];
  const float* b1     = (const float*)d_in[3];
  const float* W2     = (const float*)d_in[4];
  const float* b2     = (const float*)d_in[5];

  float* out  = (float*)d_out;
  float* traj = out + TSTEPS;  // [TSTEPS][BATCH][DIM]

  // workspace carve-up (~43 MB)
  char* p = (char*)d_ws;
  _Float16* W1t = (_Float16*)p; p += (size_t)DIM * HIDDEN * sizeof(_Float16);   // [HIDDEN][DIM]
  _Float16* W2t = (_Float16*)p; p += (size_t)HIDDEN * DIM * sizeof(_Float16);   // [DIM][HIDDEN]
  _Float16* yh  = (_Float16*)p; p += (size_t)BD * sizeof(_Float16);             // f16 GEMM1 input
  _Float16* Hh  = (_Float16*)p; p += (size_t)BATCH * HIDDEN * sizeof(_Float16); // tanh hidden, f16
  float* kbuf[4];
  for (int i = 0; i < 4; ++i) { kbuf[i] = (float*)p; p += (size_t)BD * sizeof(float); }

  const int TPB = 256;
  const int nEW = (BD + TPB - 1) / TPB;

  prep_weight_kernel<<<((size_t)DIM * HIDDEN + TPB - 1) / TPB, TPB, 0, stream>>>(
      W1, W1t, DIM, HIDDEN);
  prep_weight_kernel<<<((size_t)HIDDEN * DIM + TPB - 1) / TPB, TPB, 0, stream>>>(
      W2, W2t, HIDDEN, DIM);
  init_kernel<<<nEW, TPB, 0, stream>>>(y_init, t_span, out, yh);

  const dim3 blk(256);
  const dim3 g1(HIDDEN / 128, BATCH / 128);  // GEMM1: 32x8 = 256 blocks
  const dim3 g2(DIM / 64, BATCH / 64);       // GEMM2: 16x16 = 256 blocks

  for (int s = 0; s < TSTEPS - 1; ++s) {
    const float* yprev = traj + (size_t)s * BD;
    float*       ynext = traj + (size_t)(s + 1) * BD;

    // k1 = f(y)
    gemm_wmma_kernel<128, 128, 2, 4, true ><<<g1, blk, 0, stream>>>(yh, W1t, b1, Hh, BATCH, HIDDEN, DIM);
    gemm_wmma_kernel< 64,  64, 1, 2, false><<<g2, blk, 0, stream>>>(Hh, W2t, b2, kbuf[0], BATCH, DIM, HIDDEN);
    stage_axpy_kernel<<<nEW, TPB, 0, stream>>>(yprev, kbuf[0], t_span, s, 0.5f, yh);
    // k2 = f(y + h/2 k1)
    gemm_wmma_kernel<128, 128, 2, 4, true ><<<g1, blk, 0, stream>>>(yh, W1t, b1, Hh, BATCH, HIDDEN, DIM);
    gemm_wmma_kernel< 64,  64, 1, 2, false><<<g2, blk, 0, stream>>>(Hh, W2t, b2, kbuf[1], BATCH, DIM, HIDDEN);
    stage_axpy_kernel<<<nEW, TPB, 0, stream>>>(yprev, kbuf[1], t_span, s, 0.5f, yh);
    // k3 = f(y + h/2 k2)
    gemm_wmma_kernel<128, 128, 2, 4, true ><<<g1, blk, 0, stream>>>(yh, W1t, b1, Hh, BATCH, HIDDEN, DIM);
    gemm_wmma_kernel< 64,  64, 1, 2, false><<<g2, blk, 0, stream>>>(Hh, W2t, b2, kbuf[2], BATCH, DIM, HIDDEN);
    stage_axpy_kernel<<<nEW, TPB, 0, stream>>>(yprev, kbuf[2], t_span, s, 1.0f, yh);
    // k4 = f(y + h k3)
    gemm_wmma_kernel<128, 128, 2, 4, true ><<<g1, blk, 0, stream>>>(yh, W1t, b1, Hh, BATCH, HIDDEN, DIM);
    gemm_wmma_kernel< 64,  64, 1, 2, false><<<g2, blk, 0, stream>>>(Hh, W2t, b2, kbuf[3], BATCH, DIM, HIDDEN);
    // y_{s+1}
    rk4_update_kernel<<<nEW, TPB, 0, stream>>>(yprev, kbuf[0], kbuf[1], kbuf[2],
                                               kbuf[3], t_span, s, ynext, yh);
  }
}
